// LSTM_50062138802237
// MI455X (gfx1250) — matrix operand order repaired and checked by reading, not compiled, not run on hardware
//
#include <hip/hip_runtime.h>
#include <hip/hip_bf16.h>

// ---------------------------------------------------------------------------
// Bidirectional 2-layer LSTM on MI455X (gfx1250, wave32, WMMA).
//  - f16 data / f32 accumulate via v_wmma_f32_16x16x32_f16
//  - gx (input projections) precomputed by a parallel WMMA GEMM (8 M-tiles
//    per block, B-fragments register-reused, A staged via async-to-LDS),
//    stored in WMMA accumulator-fragment order, streamed with NT hints.
//  - recurrent scan: 16 batch rows per block, w_hh register-resident as
//    16 prebuilt B-fragments, h chained through LDS (no global sync),
//    WMMA chain independent of the gx global load (gx folded in epilogue).
// Workspace use: ~421 MB.
// ---------------------------------------------------------------------------

typedef __attribute__((ext_vector_type(16))) _Float16 v16h;
typedef __attribute__((ext_vector_type(8)))  _Float16 v8h;
typedef __attribute__((ext_vector_type(8)))  float    v8f;

#define DEV __device__ __forceinline__

constexpr int B_ = 256, S_ = 512, E_ = 300, EP = 320, H_ = 128, G_ = 512; // G = 4H
constexpr int NB = B_ * S_;          // 131072 rows
constexpr int MT = NB / 16;          // 8192 M-tiles (16 rows each)

#if defined(__has_builtin)
#if __has_builtin(__builtin_amdgcn_tanhf)
#define TANH_FAST 1
#endif
#endif

DEV float tanh_(float x) {
#ifdef TANH_FAST
  return __builtin_amdgcn_tanhf(x);      // v_tanh_f32
#else
  float e = __expf(2.0f * x);
  return (e - 1.0f) / (e + 1.0f);
#endif
}
DEV float sigm_(float x) { return 0.5f * tanh_(0.5f * x) + 0.5f; }

// --- f32 -> f16 weight conversion with optional K zero-padding -------------
__global__ void cvt_pad_f16(const float* __restrict__ src, _Float16* __restrict__ dst,
                            int srcK, int dstK) {
  int r = blockIdx.x;
  for (int j = threadIdx.x; j < dstK; j += blockDim.x) {
    float v = (j < srcK) ? src[(size_t)r * srcK + j] : 0.0f;
    dst[(size_t)r * dstK + j] = (_Float16)v;
  }
}

// --- embedding gather: tokens[B,S] -> xh[S,B,EP] (f16, zero-padded) --------
__global__ void embed_kernel(const int* __restrict__ tokens, const float* __restrict__ emb,
                             _Float16* __restrict__ xh) {
  int row = blockIdx.x;                 // row = s*256 + b  (time-major)
  int s = row >> 8, b = row & 255;
  int tok = tokens[b * S_ + s];
  const float* e = emb + (size_t)tok * E_;
  _Float16* o = xh + (size_t)row * EP;
  for (int j = threadIdx.x; j < EP; j += blockDim.x) {
    _Float16 v = (_Float16)((j < E_) ? e[j] : 0.0f);
    __builtin_nontemporal_store(v, o + j);
  }
}

// --- gx GEMM: [NB,Kp] x [512,Kp]^T + bias -> fragment-ordered f16 ----------
// Block = 8 M-tiles (128 contiguous rows), full N=512. Wave w owns n-tiles
// {g*8+w}: gates (i,f,g,o) for hidden slice [w*16, w*16+16) -- identical
// mapping to the scan kernel, so gx is stored exactly as the scan reads it:
// gx[dir][mt][wave][gate][lane][8 halves].
template <int KP>
__global__ __launch_bounds__(256, 1)
void gemm_gx(const _Float16* __restrict__ A,     // [MT*16, KP]
             const _Float16* __restrict__ Wt,    // [2][512][KP]  (row n = K-contig)
             const float*    __restrict__ bias,  // [2][512]
             _Float16*       __restrict__ gx) {  // [2][MT][8][4][32][8]
  constexpr int KC = KP / 32;
  constexpr int MB = 8;
  int mt0 = blockIdx.x * MB;
  int d   = blockIdx.y;
  int tid = threadIdx.x;
  int lane = tid & 31, w = tid >> 5;
  int nl = lane & 15, hi = lane >> 4;

  __shared__ __align__(16) _Float16 ldsA[MB * 16 * KP];

  // Async-stage the contiguous 128 x KP A block into LDS (ASYNCcnt-tracked).
  {
    const char* src = (const char*)(A + (size_t)mt0 * 16 * KP);
    uint32_t ldsbase = (uint32_t)(uintptr_t)(&ldsA[0]);   // low 32 bits = LDS offset
    constexpr int BYTES = MB * 16 * KP * 2;
    for (int i = tid * 16; i < BYTES; i += 256 * 16) {
      uint32_t l = ldsbase + (uint32_t)i;
      uint64_t g = (uint64_t)(uintptr_t)(src + i);
      asm volatile("global_load_async_to_lds_b128 %0, %1, off"
                   :: "v"(l), "v"(g) : "memory");
    }
    asm volatile("s_wait_asynccnt 0x0" ::: "memory");
  }
  __syncthreads();

  const _Float16* W = Wt + (size_t)d * G_ * KP;

  v8f acc[MB][4];
#pragma unroll
  for (int g = 0; g < 4; ++g) {
    int nt = g * 8 + w;
    float bv = bias[d * G_ + nt * 16 + nl];
#pragma unroll
    for (int mb = 0; mb < MB; ++mb)
#pragma unroll
      for (int r = 0; r < 8; ++r) acc[mb][g][r] = bv;
  }

  for (int kc = 0; kc < KC; ++kc) {
    // B fragments once per k-chunk, reused by 8 A tiles (32 indep. WMMAs)
    v16h bf[4];
#pragma unroll
    for (int g = 0; g < 4; ++g) {
      int nt = g * 8 + w;
      bf[g] = *(const v16h*)(W + (size_t)(nt * 16 + nl) * KP + kc * 32 + hi * 16);
    }
#pragma unroll
    for (int mb = 0; mb < MB; ++mb) {
      union { v16h v; v8h h[2]; } af;
      const _Float16* ap = ldsA + (mb * 16 + nl) * KP + kc * 32 + hi * 16;
      af.h[0] = *(const v8h*)(ap);
      af.h[1] = *(const v8h*)(ap + 8);
#pragma unroll
      for (int g = 0; g < 4; ++g)
        acc[mb][g] = __builtin_amdgcn_wmma_f32_16x16x32_f16(false, af.v, false, bf[g],
                                                            (short)0, acc[mb][g], false, false);
    }
  }

#pragma unroll
  for (int mb = 0; mb < MB; ++mb)
#pragma unroll
    for (int g = 0; g < 4; ++g) {
      v8h o;
#pragma unroll
      for (int r = 0; r < 8; ++r) o[r] = (_Float16)acc[mb][g][r];
      _Float16* dst = gx + (size_t)d * MT * 8192
                         + ((((size_t)(mt0 + mb) * 8 + w) * 4 + g) * 32 + lane) * 8;
      __builtin_nontemporal_store(o, (v8h*)dst);  // streamed: consumed once, later
    }
}

// --- serial LSTM scan: grid (16 batch-tiles, 2 dirs), 8 waves --------------
template <int WRITE_SEQ>
__global__ __launch_bounds__(256, 1)
void lstm_scan(const _Float16* __restrict__ gx_all,  // [2][MT][8][4][32][8]
               const _Float16* __restrict__ whh,     // [2][512][128]
               _Float16*       __restrict__ h_seq,   // [S][256][256] or null
               float*          __restrict__ hT) {    // [2][256][128] or null
  int bt = blockIdx.x;                 // batch rows [bt*16, bt*16+16)
  int d  = blockIdx.y;                 // 0 = forward, 1 = backward
  int tid = threadIdx.x, lane = tid & 31, w = tid >> 5;
  int nl = lane & 15, hi = lane >> 4;

  __shared__ __align__(16) _Float16 lds_h[16 * 136]; // [batch row][hidden k], padded
  for (int i = tid; i < 16 * 136; i += 256) lds_h[i] = (_Float16)0.f;
  __syncthreads();

  // Register-resident B fragments of w_hh^T (loop-invariant across 512 steps).
  v16h bfrag[4][4];
  const _Float16* Wd = whh + (size_t)d * G_ * H_;
#pragma unroll
  for (int g = 0; g < 4; ++g) {
    int n = (g * 8 + w) * 16 + nl;
#pragma unroll
    for (int kc = 0; kc < 4; ++kc)
      bfrag[g][kc] = *(const v16h*)(Wd + (size_t)n * H_ + kc * 32 + hi * 16);
  }

  const _Float16* gxd = gx_all + (size_t)d * MT * 8192;
  int col = w * 16 + nl;               // hidden index owned by this lane
  float cst[8], h8[8];
#pragma unroll
  for (int r = 0; r < 8; ++r) { cst[r] = 0.f; h8[r] = 0.f; }

  for (int t = 0; t < S_; ++t) {
    int s  = d ? (S_ - 1 - t) : t;
    int mt = s * 16 + bt;

    // Issue gx loads early (NT); they overlap the whole WMMA chain below.
    v8h gxv[4];
#pragma unroll
    for (int g = 0; g < 4; ++g)
      gxv[g] = __builtin_nontemporal_load(
          (const v8h*)(gxd + ((((size_t)mt * 8 + w) * 4 + g) * 32 + lane) * 8));
    if (t + 1 < S_) {                  // prefetch next step's gx tile
      int s2 = d ? (S_ - 2 - t) : (t + 1);
      __builtin_prefetch(gxd + ((((size_t)(s2 * 16 + bt) * 8 + w) * 4) * 32 + lane) * 8, 0, 0);
    }

    // A fragments from previous h (LDS) -- the only WMMA dependency
    union { v16h v; v8h h[2]; } af[4];
#pragma unroll
    for (int kc = 0; kc < 4; ++kc) {
      const _Float16* ap = lds_h + nl * 136 + kc * 32 + hi * 16;
      af[kc].h[0] = *(const v8h*)(ap);
      af[kc].h[1] = *(const v8h*)(ap + 8);
    }

    // z_rec = h_prev @ w_hh^T : 16 WMMAs starting from zero accumulators
    v8f acc[4];
#pragma unroll
    for (int g = 0; g < 4; ++g)
#pragma unroll
      for (int r = 0; r < 8; ++r) acc[g][r] = 0.f;
#pragma unroll
    for (int kc = 0; kc < 4; ++kc)
#pragma unroll
      for (int g = 0; g < 4; ++g)
        acc[g] = __builtin_amdgcn_wmma_f32_16x16x32_f16(false, af[kc].v, false,
                       bfrag[g][kc], (short)0, acc[g], false, false);

    __syncthreads();  // all waves done reading h_prev before we overwrite it

    // gates (gx folded here); each lane owns all 4 gates of its 8 cells
#pragma unroll
    for (int r = 0; r < 8; ++r) {
      float iv = sigm_(acc[0][r] + (float)gxv[0][r]);
      float fv = sigm_(acc[1][r] + (float)gxv[1][r]);
      float gv = tanh_(acc[2][r] + (float)gxv[2][r]);
      float ov = sigm_(acc[3][r] + (float)gxv[3][r]);
      float c  = fv * cst[r] + iv * gv;
      cst[r] = c;
      float h = ov * tanh_(c);
      h8[r] = h;
      int m = r + hi * 8;
      lds_h[m * 136 + col] = (_Float16)h;
      if (WRITE_SEQ)
        h_seq[((size_t)s * 256 + bt * 16 + m) * 256 + d * 128 + col] = (_Float16)h;
    }
    __syncthreads(); // h(t) visible before next step's A reads
  }

  if (!WRITE_SEQ) {
#pragma unroll
    for (int r = 0; r < 8; ++r) {
      int m = r + hi * 8;
      hT[((size_t)d * 256 + bt * 16 + m) * 128 + col] = h8[r];
    }
  }
}

// --- tiny FC head: relu(hT @ fc1^T + b1) @ fc2^T + b2 ----------------------
__global__ void head_kernel(const float* __restrict__ hT,   // [2][256][128]
                            const float* __restrict__ fc1w, const float* __restrict__ fc1b,
                            const float* __restrict__ fc2w, const float* __restrict__ fc2b,
                            float* __restrict__ out) {
  int b = blockIdx.x * blockDim.x + threadIdx.x;
  if (b >= 256) return;
  const float* hf = hT + (size_t)b * 128;
  const float* hb = hT + (size_t)(256 + b) * 128;
  float o0 = fc2b[0], o1 = fc2b[1];
  for (int j = 0; j < 128; ++j) {
    float s = fc1b[j];
    const float* wr = fc1w + (size_t)j * 256;
    for (int k = 0; k < 128; ++k) s += hf[k] * wr[k];
    for (int k = 0; k < 128; ++k) s += hb[k] * wr[128 + k];
    s = fmaxf(s, 0.f);
    o0 += s * fc2w[j];
    o1 += s * fc2w[128 + j];
  }
  out[b * 2 + 0] = o0;
  out[b * 2 + 1] = o1;
}

extern "C" void kernel_launch(void* const* d_in, const int* in_sizes, int n_in,
                              void* d_out, int out_size, void* d_ws, size_t ws_size,
                              hipStream_t stream) {
  (void)in_sizes; (void)n_in; (void)out_size; (void)ws_size;
  const int*   tokens = (const int*)d_in[0];
  const float* emb    = (const float*)d_in[1];
  const float* w_ih0  = (const float*)d_in[2];
  const float* w_hh0  = (const float*)d_in[3];
  const float* b0     = (const float*)d_in[4];
  const float* w_ih1  = (const float*)d_in[5];
  const float* w_hh1  = (const float*)d_in[6];
  const float* b1     = (const float*)d_in[7];
  const float* fc1w   = (const float*)d_in[8];
  const float* fc1b   = (const float*)d_in[9];
  const float* fc2w   = (const float*)d_in[10];
  const float* fc2b   = (const float*)d_in[11];
  float* out = (float*)d_out;

  char* ws = (char*)d_ws;
  size_t off = 0;
  auto alloc = [&](size_t bytes) -> void* {
    void* p = ws + off;
    off = (off + bytes + 255) & ~(size_t)255;
    return p;
  };
  _Float16* xh    = (_Float16*)alloc((size_t)NB * EP * 2);         //  84 MB
  _Float16* gx    = (_Float16*)alloc((size_t)2 * NB * 512 * 2);    // 268 MB (reused L0/L1)
  _Float16* hseq  = (_Float16*)alloc((size_t)NB * 256 * 2);        //  67 MB
  _Float16* wih0h = (_Float16*)alloc((size_t)2 * 512 * EP * 2);
  _Float16* wih1h = (_Float16*)alloc((size_t)2 * 512 * 256 * 2);
  _Float16* whh0h = (_Float16*)alloc((size_t)2 * 512 * 128 * 2);
  _Float16* whh1h = (_Float16*)alloc((size_t)2 * 512 * 128 * 2);
  float*    hT    = (float*)alloc((size_t)2 * 256 * 128 * 4);

  // weight prep (f32 -> f16, pad E 300->320)
  cvt_pad_f16<<<1024, 256, 0, stream>>>(w_ih0, wih0h, 300, 320);
  cvt_pad_f16<<<1024, 256, 0, stream>>>(w_ih1, wih1h, 256, 256);
  cvt_pad_f16<<<1024, 256, 0, stream>>>(w_hh0, whh0h, 128, 128);
  cvt_pad_f16<<<1024, 256, 0, stream>>>(w_hh1, whh1h, 128, 128);

  embed_kernel<<<NB, 64, 0, stream>>>(tokens, emb, xh);

  // layer 0
  gemm_gx<320><<<dim3(MT / 8, 2), 256, 0, stream>>>(xh, wih0h, b0, gx);
  lstm_scan<1><<<dim3(16, 2), 256, 0, stream>>>(gx, whh0h, hseq, nullptr);
  // layer 1 (gx buffer reused)
  gemm_gx<256><<<dim3(MT / 8, 2), 256, 0, stream>>>(hseq, wih1h, b1, gx);
  lstm_scan<0><<<dim3(16, 2), 256, 0, stream>>>(gx, whh1h, nullptr, hT);

  head_kernel<<<1, 256, 0, stream>>>(hT, fc1w, fc1b, fc2w, fc2b, out);
}
